// HybridCrossAttnProcessor_48524540510829
// MI455X (gfx1250) — matrix operand order repaired and checked by reading, not compiled, not run on hardware
//
#include <hip/hip_runtime.h>

// ---------------------------------------------------------------------------
// CDNA5 (gfx1250) sparse top-k cross-attention, bf16 WMMA everywhere.
// B=2, Q=2048, KV=16384, C=1024, H=16, D=64, TOPK=2048.
// ---------------------------------------------------------------------------

typedef __attribute__((ext_vector_type(16))) __bf16 v16bf;
typedef __attribute__((ext_vector_type(8)))  float  v8f;

__device__ __forceinline__ v8f vzero8() {
  v8f v;
#pragma unroll
  for (int i = 0; i < 8; ++i) v[i] = 0.0f;
  return v;
}

__device__ __forceinline__ v8f wmma_bf16(v16bf a, v16bf b, v8f c) {
  // D = A(16x32 bf16) x B(32x16 bf16) + C(16x16 f32)
  return __builtin_amdgcn_wmma_f32_16x16x32_bf16(
      /*neg_a=*/false, a, /*neg_b=*/false, b,
      /*c_mod=*/(short)0, c, /*reuse_a=*/false, /*reuse_b=*/false);
}

// A fragment, 16x32 bf16, row-major tile with leading dim `ld` (contiguous:
// lowers to 2x ds_load_b128 per lane).
// lane L holds row M=L&15; lanes 0-15: K=0..7,16..23 ; lanes 16-31: K=8..15,24..31
__device__ __forceinline__ v16bf frag_a16x32(const __bf16* tile, int ld) {
  int lane = threadIdx.x & 31;
  const __bf16* rp = tile + (size_t)(lane & 15) * ld + ((lane >> 4) ? 8 : 0);
  v16bf a;
#pragma unroll
  for (int e = 0; e < 8; ++e) a[e] = rp[e];
#pragma unroll
  for (int e = 8; e < 16; ++e) a[e] = rp[e + 8];
  return a;
}

// B fragment, 32x16 bf16, tile stored COLUMN-major: tile[n*ld + k]
// (contiguous per lane: 2x ds_load_b128).
// lane L holds col N=L&15; lanes 0-15: K=0..7,16..23 ; lanes 16-31: K=8..15,24..31
__device__ __forceinline__ v16bf frag_b32x16_cm(const __bf16* tile, int ld) {
  int lane = threadIdx.x & 31;
  const __bf16* rp = tile + (size_t)(lane & 15) * ld + ((lane >> 4) ? 8 : 0);
  v16bf b;
#pragma unroll
  for (int e = 0; e < 8; ++e) b[e] = rp[e];
#pragma unroll
  for (int e = 8; e < 16; ++e) b[e] = rp[e + 8];
  return b;
}

// ---------------------------------------------------------------------------
// Tiled GEMM: Out[M,N] = A[M,K] * W[K,N]  (+ fused epilogues)
// Block tile 128x128, K-slice 32, double-buffered LDS (global loads for the
// next stage issued before this stage's WMMAs). 256 threads = 8 waves, each
// wave computes 32x64 (8 f32 accumulators).
// mode 0: Out = acc
// mode 1: Out = extra * sigmoid(acc + bias[n])       (gate)
// mode 2: Out = acc + bias[n] + extra                (out-proj + residual)
// ---------------------------------------------------------------------------
struct TileRegs { float a[16]; float b[16]; };

__device__ __forceinline__ void load_tile_g2r(const float* __restrict__ A,
                                              const float* __restrict__ W,
                                              int K, int N, int bm, int bn,
                                              int k0, int tid, TileRegs& t) {
  {
    int r = tid >> 1, c0 = (tid & 1) * 16;
    const float4* src =
        reinterpret_cast<const float4*>(A + (size_t)(bm + r) * K + k0 + c0);
#pragma unroll
    for (int v = 0; v < 4; ++v) {
      float4 q = src[v];
      t.a[v * 4 + 0] = q.x; t.a[v * 4 + 1] = q.y;
      t.a[v * 4 + 2] = q.z; t.a[v * 4 + 3] = q.w;
    }
  }
  {
    int r = tid >> 3, c0 = (tid & 7) * 16;
    const float4* src =
        reinterpret_cast<const float4*>(W + (size_t)(k0 + r) * N + bn + c0);
#pragma unroll
    for (int v = 0; v < 4; ++v) {
      float4 q = src[v];
      t.b[v * 4 + 0] = q.x; t.b[v * 4 + 1] = q.y;
      t.b[v * 4 + 2] = q.z; t.b[v * 4 + 3] = q.w;
    }
  }
}

__device__ __forceinline__ void store_tile_r2l(__bf16 (*As)[32], __bf16 (*BsT)[32],
                                               int tid, const TileRegs& t) {
  int ra = tid >> 1, ca = (tid & 1) * 16;
#pragma unroll
  for (int e = 0; e < 16; ++e) As[ra][ca + e] = (__bf16)t.a[e];  // contiguous
  int rb = tid >> 3, cb = (tid & 7) * 16;
#pragma unroll
  for (int e = 0; e < 16; ++e) BsT[cb + e][rb] = (__bf16)t.b[e]; // transpose
}

__global__ void __launch_bounds__(256)
gemm_wmma_kernel(const float* __restrict__ A, const float* __restrict__ W,
                 float* __restrict__ Out, int M, int K, int N, int mode,
                 const float* __restrict__ bias, const float* __restrict__ extra) {
  __shared__ __bf16 As[2][128][32];   // activations, row-major
  __shared__ __bf16 BsT[2][128][32];  // weights, column-major (BsT[n][k])

  int tid  = threadIdx.x;
  int wave = tid >> 5;
  int lane = tid & 31;
  int bm = blockIdx.x * 128;
  int bn = blockIdx.y * 128;
  int wm = (wave & 3) * 32;   // 0,32,64,96
  int wn = (wave >> 2) * 64;  // 0,64

  v8f acc[2][4];
#pragma unroll
  for (int mt = 0; mt < 2; ++mt)
#pragma unroll
    for (int nt = 0; nt < 4; ++nt) acc[mt][nt] = vzero8();

  TileRegs tr;
  load_tile_g2r(A, W, K, N, bm, bn, 0, tid, tr);
  store_tile_r2l(As[0], BsT[0], tid, tr);

  int buf = 0;
  for (int k0 = 0; k0 < K; k0 += 32) {
    __syncthreads();
    bool nxt = (k0 + 32) < K;
    if (nxt) load_tile_g2r(A, W, K, N, bm, bn, k0 + 32, tid, tr);
#pragma unroll
    for (int mt = 0; mt < 2; ++mt) {
      v16bf af = frag_a16x32(&As[buf][wm + mt * 16][0], 32);
#pragma unroll
      for (int nt = 0; nt < 4; ++nt) {
        v16bf bf = frag_b32x16_cm(&BsT[buf][wn + nt * 16][0], 32);
        acc[mt][nt] = wmma_bf16(af, bf, acc[mt][nt]);
      }
    }
    if (nxt) store_tile_r2l(As[buf ^ 1], BsT[buf ^ 1], tid, tr);
    buf ^= 1;
  }

  int mr = (lane >> 4) << 3;  // row offset 0 or 8 within 16x16 tile
#pragma unroll
  for (int mt = 0; mt < 2; ++mt)
#pragma unroll
    for (int nt = 0; nt < 4; ++nt)
#pragma unroll
      for (int r = 0; r < 8; ++r) {
        int row = bm + wm + mt * 16 + mr + r;
        int col = bn + wn + nt * 16 + (lane & 15);
        size_t off = (size_t)row * N + col;
        float v = acc[mt][nt][r];
        if (mode == 1) {
          float g = v + bias[col];
          Out[off] = extra[off] * (1.0f / (1.0f + __expf(-g)));
        } else if (mode == 2) {
          Out[off] = v + bias[col] + extra[off];
        } else {
          Out[off] = v;
        }
      }
}

// ---------------------------------------------------------------------------
// q_summary = mean over Q of q projections: [B, C]
// ---------------------------------------------------------------------------
__global__ void __launch_bounds__(256)
qsum_kernel(const float* __restrict__ qbuf, float* __restrict__ qsum,
            int Qn, int C) {
  int c = blockIdx.x * blockDim.x + threadIdx.x;
  int b = blockIdx.y;
  if (c >= C) return;
  float s = 0.f;
  const float* p = qbuf + (size_t)b * Qn * C + c;
  for (int q = 0; q < Qn; ++q) s += p[(size_t)q * C];
  qsum[b * C + c] = s / (float)Qn;
}

// ---------------------------------------------------------------------------
// Selection scores -> monotone uint keys: key[b,h,kv]
// ---------------------------------------------------------------------------
__global__ void __launch_bounds__(256)
score_kernel(const float* __restrict__ kbuf, const float* __restrict__ qsum,
             unsigned* __restrict__ keys, int KV, int C, int D, float scale) {
  int kv = blockIdx.x * blockDim.x + threadIdx.x;
  int h = blockIdx.y, b = blockIdx.z, H = gridDim.y;
  if (kv >= KV) return;
  const float* kr = kbuf + ((size_t)b * KV + kv) * C + h * D;
  const float* qs = qsum + (size_t)b * C + h * D;
  float s = 0.f;
#pragma unroll 8
  for (int d = 0; d < 64; ++d) s += kr[d] * qs[d];
  s *= scale;
  unsigned u = __float_as_uint(s);
  u = (u & 0x80000000u) ? ~u : (u | 0x80000000u);  // monotone transform
  keys[((size_t)(b * H + h)) * KV + kv] = u;
}

// ---------------------------------------------------------------------------
// Radix-select top-K indices per (b,h). Softmax over the selected set is
// permutation invariant, so unordered selection is sufficient.
// ---------------------------------------------------------------------------
__global__ void __launch_bounds__(256)
topk_kernel(const unsigned* __restrict__ keys, int* __restrict__ idx_out,
            int KV, int TOPKN) {
  __shared__ unsigned hist[256];
  __shared__ unsigned sh_prefix, sh_want;
  __shared__ int cnt_gt, cnt_eq;
  const unsigned* kk = keys + (size_t)blockIdx.x * KV;
  int tid = threadIdx.x;
  if (tid == 0) { sh_prefix = 0u; sh_want = (unsigned)TOPKN; }
  __syncthreads();

  for (int pass = 0; pass < 4; ++pass) {
    int shift = 24 - pass * 8;
    hist[tid] = 0u;
    __syncthreads();
    unsigned prefix = sh_prefix;
    unsigned mask = (pass == 0) ? 0u : (0xFFFFFFFFu << (shift + 8));
    for (int i = tid; i < KV; i += 256) {
      unsigned u = kk[i];
      if ((u & mask) == prefix) atomicAdd(&hist[(u >> shift) & 255u], 1u);
    }
    __syncthreads();
    if (tid == 0) {
      unsigned want = sh_want, cum = 0; int sel = 0;
      for (int bnum = 255; bnum >= 0; --bnum) {
        unsigned c = hist[bnum];
        if (cum + c >= want) { sel = bnum; break; }
        cum += c;
      }
      sh_prefix = prefix | ((unsigned)sel << shift);
      sh_want = want - cum;
    }
    __syncthreads();
  }

  unsigned T = sh_prefix;
  if (tid == 0) { cnt_gt = 0; cnt_eq = 0; }
  __syncthreads();
  int* out = idx_out + (size_t)blockIdx.x * TOPKN;
  for (int i = tid; i < KV; i += 256)
    if (kk[i] > T) out[atomicAdd(&cnt_gt, 1)] = i;
  __syncthreads();
  int base = cnt_gt;
  for (int i = tid; i < KV; i += 256)
    if (kk[i] == T) {
      int p = atomicAdd(&cnt_eq, 1);
      if (base + p < TOPKN) out[base + p] = i;
    }
}

// ---------------------------------------------------------------------------
// Flash attention over the selected keys.
// Block: 128 threads (4 waves), one (b, h, 64-query tile) per block.
// Per-chunk: gather 32 keys via idx list, S = Q K^T (WMMA), online softmax,
// O += P V (WMMA). Accumulators in registers; K key-major / V d-major in LDS
// so both B fragments are lane-contiguous (ds_load_b128 path).
// ---------------------------------------------------------------------------
__global__ void __launch_bounds__(128)
attn_kernel(const float* __restrict__ qbuf, const float* __restrict__ kbuf,
            const float* __restrict__ vbuf, const int* __restrict__ idx,
            float* __restrict__ obuf, int Qn, int KV, int C, int TOPKN,
            float scale) {
  __shared__ __bf16 Qs[64][64];   // q tile (row-major)
  __shared__ __bf16 Ks[32][64];   // keys chunk, key-major -> B frag for QK^T
  __shared__ __bf16 VTs[64][32];  // values chunk, d-major -> B frag for PV
  __shared__ float  Ss[4][16][32];
  __shared__ __bf16 Ps[4][16][32];
  __shared__ float  fac[64];

  int tid = threadIdx.x, wave = tid >> 5, lane = tid & 31;
  int q0 = blockIdx.x * 64;
  int h = blockIdx.y, b = blockIdx.z, H = gridDim.y;
  int row0 = wave * 16;
  int mr = (lane >> 4) << 3;

  for (int i = tid; i < 64 * 64; i += 128) {
    int r = i >> 6, d = i & 63;
    Qs[r][d] = (__bf16)qbuf[((size_t)b * Qn + q0 + r) * C + h * 64 + d];
  }

  float m_r = -1e30f, l_r = 0.f;
  v8f o[4];
#pragma unroll
  for (int t = 0; t < 4; ++t) o[t] = vzero8();

  const int* il = idx + (size_t)(b * H + h) * TOPKN;

  for (int j = 0; j < TOPKN; j += 32) {
    __syncthreads();
    for (int i = tid; i < 32 * 64; i += 128) {  // gather K/V chunk
      int kki = i >> 6, d = i & 63;
      int g = il[j + kki];
      size_t src = ((size_t)b * KV + g) * C + h * 64 + d;
      Ks[kki][d]  = (__bf16)kbuf[src];
      VTs[d][kki] = (__bf16)vbuf[src];
    }
    __syncthreads();

    // S(16x32) = Q(16x64) x K^T(64x32)
    v8f s0 = vzero8(), s1 = vzero8();
#pragma unroll
    for (int kk = 0; kk < 64; kk += 32) {
      v16bf af = frag_a16x32(&Qs[row0][kk], 64);
      v16bf b0 = frag_b32x16_cm(&Ks[0][kk], 64);    // B[k=d][n=key]
      v16bf b1 = frag_b32x16_cm(&Ks[16][kk], 64);
      s0 = wmma_bf16(af, b0, s0);
      s1 = wmma_bf16(af, b1, s1);
    }
#pragma unroll
    for (int r = 0; r < 8; ++r) {
      Ss[wave][mr + r][lane & 15]        = s0[r] * scale;
      Ss[wave][mr + r][(lane & 15) + 16] = s1[r] * scale;
    }

    // online softmax: lanes 0-15 each own one row (same-wave DS ops in order)
    if (lane < 16) {
      float cmax = -1e30f;
#pragma unroll
      for (int e = 0; e < 32; ++e) cmax = fmaxf(cmax, Ss[wave][lane][e]);
      float newm = fmaxf(m_r, cmax);
      float alpha = __expf(m_r - newm);
      float sum = 0.f;
#pragma unroll
      for (int e = 0; e < 32; ++e) {
        float p = __expf(Ss[wave][lane][e] - newm);
        sum += p;
        Ps[wave][lane][e] = (__bf16)p;
      }
      l_r = l_r * alpha + sum;
      m_r = newm;
      fac[row0 + lane] = alpha;
    }

    float fr[8];
#pragma unroll
    for (int r = 0; r < 8; ++r) fr[r] = fac[row0 + mr + r];
#pragma unroll
    for (int t = 0; t < 4; ++t)
#pragma unroll
      for (int r = 0; r < 8; ++r) o[t][r] *= fr[r];

    // O(16x64) += P(16x32) x V(32x64)
    v16bf pa = frag_a16x32(&Ps[wave][0][0], 32);
#pragma unroll
    for (int t = 0; t < 4; ++t) {
      v16bf bv = frag_b32x16_cm(&VTs[t * 16][0], 32);  // B[k=key][n=d]
      o[t] = wmma_bf16(pa, bv, o[t]);
    }
  }

  // normalize + write out (head-interleaved [B,Q,C])
  if (lane < 16) fac[row0 + lane] = 1.0f / l_r;
  float lr[8];
#pragma unroll
  for (int r = 0; r < 8; ++r) lr[r] = fac[row0 + mr + r];
#pragma unroll
  for (int t = 0; t < 4; ++t)
#pragma unroll
    for (int r = 0; r < 8; ++r) {
      int row = q0 + row0 + mr + r;
      int col = h * 64 + t * 16 + (lane & 15);
      obuf[((size_t)b * Qn + row) * C + col] = o[t][r] * lr[r];
    }
}

// ---------------------------------------------------------------------------
extern "C" void kernel_launch(void* const* d_in, const int* in_sizes, int n_in,
                              void* d_out, int out_size, void* d_ws, size_t ws_size,
                              hipStream_t stream) {
  (void)in_sizes; (void)n_in; (void)out_size; (void)ws_size;
  const float* hidden = (const float*)d_in[0];
  const float* enc    = (const float*)d_in[1];
  const float* Wq     = (const float*)d_in[2];
  const float* Wk     = (const float*)d_in[3];
  const float* Wv     = (const float*)d_in[4];
  const float* Wo     = (const float*)d_in[5];
  const float* bo     = (const float*)d_in[6];
  const float* Wg     = (const float*)d_in[7];
  const float* bg     = (const float*)d_in[8];
  float* out = (float*)d_out;

  const int B = 2, Qn = 2048, KV = 16384, C = 1024, H = 16, D = 64, TOPKN = 2048;
  const float scale = 0.125f;  // 1/sqrt(64)

  float* qbuf = (float*)d_ws;
  float* kbuf = qbuf + (size_t)B * Qn * C;
  float* vbuf = kbuf + (size_t)B * KV * C;
  float* attn = vbuf + (size_t)B * KV * C;
  float* zbuf = attn + (size_t)B * Qn * C;
  float* qsum = zbuf + (size_t)B * Qn * C;
  unsigned* keys = (unsigned*)(qsum + (size_t)B * C);
  int* idx = (int*)(keys + (size_t)B * H * KV);

  // Projections (WMMA bf16 GEMMs)
  gemm_wmma_kernel<<<dim3((B * Qn) / 128, C / 128), 256, 0, stream>>>(
      hidden, Wq, qbuf, B * Qn, C, C, 0, nullptr, nullptr);
  gemm_wmma_kernel<<<dim3((B * KV) / 128, C / 128), 256, 0, stream>>>(
      enc, Wk, kbuf, B * KV, C, C, 0, nullptr, nullptr);
  gemm_wmma_kernel<<<dim3((B * KV) / 128, C / 128), 256, 0, stream>>>(
      enc, Wv, vbuf, B * KV, C, C, 0, nullptr, nullptr);

  // q summary (mean over Q), selection scores, top-k indices
  qsum_kernel<<<dim3(C / 256, B), 256, 0, stream>>>(qbuf, qsum, Qn, C);
  score_kernel<<<dim3(KV / 256, H, B), 256, 0, stream>>>(kbuf, qsum, keys, KV, C, D, scale);
  topk_kernel<<<dim3(B * H), 256, 0, stream>>>(keys, idx, KV, TOPKN);

  // flash attention over selected keys
  attn_kernel<<<dim3(Qn / 64, H, B), 128, 0, stream>>>(
      qbuf, kbuf, vbuf, idx, attn, Qn, KV, C, TOPKN, scale);

  // gate:  z = attn * sigmoid(attn @ Wg + bg)
  gemm_wmma_kernel<<<dim3((B * Qn) / 128, C / 128), 256, 0, stream>>>(
      attn, Wg, zbuf, B * Qn, C, C, 1, bg, attn);
  // out:  y = z @ Wo + bo + hidden
  gemm_wmma_kernel<<<dim3((B * Qn) / 128, C / 128), 256, 0, stream>>>(
      zbuf, Wo, out, B * Qn, C, C, 2, bo, hidden);
}
